// VariableSelectionNetwork_75900662054959
// MI455X (gfx1250) — matrix operand order repaired and checked by reading, not compiled, not run on hardware
//
#include <hip/hip_runtime.h>

typedef __attribute__((ext_vector_type(16))) _Float16 v16h;
typedef __attribute__((ext_vector_type(8)))  float    v8f;

#define TOK_PER_BLK 64
#define HDIM 128
#define FDIM 32
#define PITCH 136   // halves per LDS row (128 + 8 pad -> conflict-free ds_load_b128)

union U32h { uint4 q[2]; v16h v; };
union U4h  { _Float16 h[4]; uint2 u; };

static __device__ __forceinline__ float red_add32(float v) {
  #pragma unroll
  for (int m = 16; m >= 1; m >>= 1) v += __shfl_xor(v, m, 32);
  return v;
}
static __device__ __forceinline__ float red_max32(float v) {
  #pragma unroll
  for (int m = 16; m >= 1; m >>= 1) v = fmaxf(v, __shfl_xor(v, m, 32));
  return v;
}
static __device__ __forceinline__ float red_add16(float v) {   // within 16-lane half
  #pragma unroll
  for (int m = 8; m >= 1; m >>= 1) v += __shfl_xor(v, m, 32);
  return v;
}
static __device__ __forceinline__ float eluf(float z) { return z > 0.f ? z : (__expf(z) - 1.f); }
static __device__ __forceinline__ float sigm(float z) { return 1.f / (1.f + __expf(-z)); }

// one-shot fp32 -> f16 conversion of f_fc2_w into workspace
__global__ __launch_bounds__(256) void cvt_w_kernel(const float* __restrict__ src,
                                                    _Float16* __restrict__ dst) {
  int i = blockIdx.x * 256 + threadIdx.x;     // 262144 threads, 4 elems each
  float4 w4 = ((const float4*)src)[i];
  U4h pk;
  pk.h[0] = (_Float16)w4.x; pk.h[1] = (_Float16)w4.y;
  pk.h[2] = (_Float16)w4.z; pk.h[3] = (_Float16)w4.w;
  ((uint2*)dst)[i] = pk.u;
}

template <bool PRECONV>
__global__ __launch_bounds__(256) void vsn_kernel(
    const float* __restrict__ x,
    const float* __restrict__ wg1w, const float* __restrict__ wg1b,
    const float* __restrict__ wg2w, const float* __restrict__ wg2b,
    const float* __restrict__ wlng, const float* __restrict__ wlnb,
    const float* __restrict__ f1w,  const float* __restrict__ f1b,
    const float* __restrict__ f2w,  const float* __restrict__ f2b,
    const float* __restrict__ fsw,  const float* __restrict__ fsb,
    const float* __restrict__ flng, const float* __restrict__ flnb,
    const _Float16* __restrict__ w16,   // pre-converted fc2 weights (PRECONV only)
    float* __restrict__ out)
{
  __shared__ alignas(16) _Float16 sW2[256 * PITCH];          // fc2 weights (f16), row g, col h
  __shared__ alignas(16) _Float16 sH[TOK_PER_BLK * PITCH];   // elu activations (f16)
  __shared__ float sX[TOK_PER_BLK * FDIM];
  __shared__ float sWgt[TOK_PER_BLK * FDIM];
  __shared__ float sP[1024];   // w1|b1|ba|bg|skw|skb|lng|lnb (128 each)
  __shared__ float sRS[2 * TOK_PER_BLK];
  __shared__ float sRQ[2 * TOK_PER_BLK];

  const int tid   = threadIdx.x;
  const int lane  = tid & 31;
  const int wv    = tid >> 5;
  const int lmod  = lane & 15;
  const int lhalf = lane >> 4;
  const int tok0  = blockIdx.x * TOK_PER_BLK;
  const int mt    = wv & 3;      // M-tile (16 tokens)
  const int nh    = wv >> 2;     // N-half (cols 0..63 / 64..127 of each GLU half)
  const int m0    = mt * 16;

  // ---- load x tile ----
  for (int i = tid; i < TOK_PER_BLK * FDIM; i += 256) sX[i] = x[(size_t)tok0 * FDIM + i];
  __syncthreads();

  // ---- Phase A: weight GRN -> softmax feature weights (lane == feature) ----
  for (int tt = 0; tt < 8; ++tt) {
    int t = wv * 8 + tt;
    float xv = sX[t * FDIM + lane];
    float acc = wg1b[lane];
    #pragma unroll
    for (int ff = 0; ff < 32; ++ff)
      acc = fmaf(__shfl(xv, ff, 32), wg1w[lane * 32 + ff], acc);
    float hv = eluf(acc);
    float a2 = wg2b[lane], g2 = wg2b[lane + 32];
    #pragma unroll
    for (int gg = 0; gg < 32; ++gg) {
      float hg = __shfl(hv, gg, 32);
      a2 = fmaf(hg, wg2w[lane * 32 + gg], a2);
      g2 = fmaf(hg, wg2w[(lane + 32) * 32 + gg], g2);
    }
    float vln  = xv + a2 * sigm(g2);
    float mean = red_add32(vln) * (1.f / 32.f);
    float d    = vln - mean;
    float var  = red_add32(d * d) * (1.f / 32.f);
    float y    = d * __frsqrt_rn(var + 1e-5f) * wlng[lane] + wlnb[lane];
    float mx   = red_max32(y);
    float e    = __expf(y - mx);
    float s    = red_add32(e);
    sWgt[t * FDIM + lane] = e / s;
  }
  __syncthreads();

  const v8f vzero = {0.f, 0.f, 0.f, 0.f, 0.f, 0.f, 0.f, 0.f};
  v8f outa[4];
  #pragma unroll
  for (int j = 0; j < 4; ++j) outa[j] = vzero;

  for (int f = 0; f < FDIM; ++f) {
    // ---- stage per-feature params ----
    if (tid < 128) {
      sP[tid]       = f1w[f * 128 + tid];
      sP[128 + tid] = f1b[f * 128 + tid];
      sP[256 + tid] = f2b[f * 256 + tid];         // bias (a half)
      sP[384 + tid] = f2b[f * 256 + 128 + tid];   // bias (gate half)
      sP[512 + tid] = fsw[f * 128 + tid];
      sP[640 + tid] = fsb[f * 128 + tid];
      sP[768 + tid] = flng[f * 128 + tid];
      sP[896 + tid] = flnb[f * 128 + tid];
    }
    __syncthreads();

    // ---- stage fc2 weights into LDS (f16) ----
    if (PRECONV) {
      // already f16 in workspace: straight B128 copy with padded LDS layout
      const uint4* wsrc = (const uint4*)(w16 + (size_t)f * 32768);
      for (int i = tid; i < 4096; i += 256) {      // 8 halves per chunk
        uint4 c4 = wsrc[i];
        int e0 = i * 8;
        int g = e0 >> 7, c = e0 & 127;
        *(uint4*)&sW2[g * PITCH + c] = c4;
      }
    } else {
      // convert fp32 -> f16 on the fly
      const float4* wsrc = (const float4*)(f2w + (size_t)f * 32768);
      for (int i = tid; i < 8192; i += 256) {
        float4 w4 = wsrc[i];
        int e0 = i * 4;
        int g = e0 >> 7, c = e0 & 127;
        U4h pk;
        pk.h[0] = (_Float16)w4.x; pk.h[1] = (_Float16)w4.y;
        pk.h[2] = (_Float16)w4.z; pk.h[3] = (_Float16)w4.w;
        *(uint2*)&sW2[g * PITCH + c] = pk.u;
      }
    }
    // ---- compute h = elu(x*w1 + b1) -> f16 LDS ----
    for (int i = tid; i < (TOK_PER_BLK * HDIM) / 4; i += 256) {
      int e0 = i * 4;
      int t = e0 >> 7, k = e0 & 127;
      float xt = sX[t * FDIM + f];
      U4h pk;
      #pragma unroll
      for (int u = 0; u < 4; ++u)
        pk.h[u] = (_Float16)eluf(fmaf(xt, sP[k + u], sP[128 + k + u]));
      *(uint2*)&sH[t * PITCH + k] = pk.u;
    }
    // prefetch next feature's weights toward L2/L1 (global_prefetch_b8)
    if (f + 1 < FDIM) {
      if (PRECONV) {
        const char* nx = (const char*)(w16 + (size_t)(f + 1) * 32768);
        for (int i = tid; i < 512; i += 256) __builtin_prefetch(nx + i * 128, 0, 0);
      } else {
        const char* nx = (const char*)(f2w + (size_t)(f + 1) * 32768);
        for (int i = tid; i < 1024; i += 256) __builtin_prefetch(nx + i * 128, 0, 0);
      }
    }
    __syncthreads();

    // ---- WMMA: (16 tok x 128) @ (128 x 64-col slice) for a-half and gate-half ----
    v8f acc[8];
    #pragma unroll
    for (int j = 0; j < 8; ++j) acc[j] = vzero;

    const int rowA = m0 + lmod;
    #pragma unroll
    for (int kt = 0; kt < 4; ++kt) {
      const int k0 = kt * 32;
      // A 16x32 f16: lane<16 holds K[k0..k0+7],[k0+16..k0+23]; lane>=16 shifted by 8
      U32h ua;
      const _Float16* pa = &sH[rowA * PITCH + k0 + 8 * lhalf];
      ua.q[0] = *(const uint4*)pa;
      ua.q[1] = *(const uint4*)(pa + 16);
      v16h av = ua.v;
      #pragma unroll
      for (int jj = 0; jj < 4; ++jj) {
        int ja = nh * 4 + jj;   // output col-tile in a half
        int jg = ja + 8;        // matching gate tile
        {
          // B 32x16 f16: lane n = column, K[k0..k0+7],[k0+16..k0+23]; lane>=16 shifted by 8
          U32h ub;
          const _Float16* pb = &sW2[(ja * 16 + lmod) * PITCH + k0 + 8 * lhalf];
          ub.q[0] = *(const uint4*)pb;
          ub.q[1] = *(const uint4*)(pb + 16);
          acc[jj] = __builtin_amdgcn_wmma_f32_16x16x32_f16(
              false, av, false, ub.v, (short)0, acc[jj], false, false);
        }
        {
          U32h ub;
          const _Float16* pb = &sW2[(jg * 16 + lmod) * PITCH + k0 + 8 * lhalf];
          ub.q[0] = *(const uint4*)pb;
          ub.q[1] = *(const uint4*)(pb + 16);
          acc[4 + jj] = __builtin_amdgcn_wmma_f32_16x16x32_f16(
              false, av, false, ub.v, (short)0, acc[4 + jj], false, false);
        }
      }
    }

    // ---- epilogue: GLU + skip + LayerNorm + weighted accumulate ----
    // C layout: vgpr v -> row m0 + v + 8*lhalf, col = tile*16 + lmod
    float xrow[8], wrow[8];
    #pragma unroll
    for (int v = 0; v < 8; ++v) {
      int m = m0 + v + 8 * lhalf;
      xrow[v] = sX[m * FDIM + f];
      wrow[v] = sWgt[m * FDIM + f];
    }
    float rs[8], rq[8];
    #pragma unroll
    for (int v = 0; v < 8; ++v) { rs[v] = 0.f; rq[v] = 0.f; }
    #pragma unroll
    for (int jj = 0; jj < 4; ++jj) {
      int n = (nh * 4 + jj) * 16 + lmod;
      float ba = sP[256 + n], bg = sP[384 + n];
      float sw = sP[512 + n], sb = sP[640 + n];
      #pragma unroll
      for (int v = 0; v < 8; ++v) {
        float a2  = acc[jj][v] + ba;
        float g2  = acc[4 + jj][v] + bg;
        float val = fmaf(a2, sigm(g2), fmaf(xrow[v], sw, sb));
        acc[jj][v] = val;
        rs[v] += val; rq[v] += val * val;
      }
    }
    #pragma unroll
    for (int v = 0; v < 8; ++v) { rs[v] = red_add16(rs[v]); rq[v] = red_add16(rq[v]); }
    // cross-wave (N-half) reduction for the LN stats
    #pragma unroll
    for (int v = 0; v < 8; ++v) {
      int m = m0 + v + 8 * lhalf;
      sRS[nh * TOK_PER_BLK + m] = rs[v];
      sRQ[nh * TOK_PER_BLK + m] = rq[v];
    }
    __syncthreads();
    float mean[8], rstd[8];
    #pragma unroll
    for (int v = 0; v < 8; ++v) {
      int m = m0 + v + 8 * lhalf;
      float ts = sRS[m] + sRS[TOK_PER_BLK + m];
      float tq = sRQ[m] + sRQ[TOK_PER_BLK + m];
      float mu = ts * (1.f / 128.f);
      float va = tq * (1.f / 128.f) - mu * mu;
      mean[v] = mu;
      rstd[v] = __frsqrt_rn(va + 1e-5f);
    }
    #pragma unroll
    for (int jj = 0; jj < 4; ++jj) {
      int n = (nh * 4 + jj) * 16 + lmod;
      float lg = sP[768 + n], lb = sP[896 + n];
      #pragma unroll
      for (int v = 0; v < 8; ++v) {
        float y = (acc[jj][v] - mean[v]) * rstd[v] * lg + lb;
        outa[jj][v] = fmaf(wrow[v], y, outa[jj][v]);
      }
    }
    __syncthreads();
  }

  // ---- store: out[token, h] ----
  #pragma unroll
  for (int jj = 0; jj < 4; ++jj) {
    int n = (nh * 4 + jj) * 16 + lmod;
    #pragma unroll
    for (int v = 0; v < 8; ++v) {
      int m = tok0 + m0 + v + 8 * lhalf;
      out[(size_t)m * HDIM + n] = outa[jj][v];
    }
  }
}

extern "C" void kernel_launch(void* const* d_in, const int* in_sizes, int n_in,
                              void* d_out, int out_size, void* d_ws, size_t ws_size,
                              hipStream_t stream) {
  (void)in_sizes; (void)n_in; (void)out_size;
  const float* x    = (const float*)d_in[0];
  const float* wg1w = (const float*)d_in[1];
  const float* wg1b = (const float*)d_in[2];
  const float* wg2w = (const float*)d_in[3];
  const float* wg2b = (const float*)d_in[4];
  const float* wlng = (const float*)d_in[5];
  const float* wlnb = (const float*)d_in[6];
  const float* f1w  = (const float*)d_in[7];
  const float* f1b  = (const float*)d_in[8];
  const float* f2w  = (const float*)d_in[9];
  const float* f2b  = (const float*)d_in[10];
  const float* fsw  = (const float*)d_in[11];
  const float* fsb  = (const float*)d_in[12];
  const float* flng = (const float*)d_in[13];
  const float* flnb = (const float*)d_in[14];
  float* out = (float*)d_out;

  const size_t w16_bytes = (size_t)FDIM * 256 * 128 * sizeof(_Float16);  // 2 MB
  dim3 grid(8192 / TOK_PER_BLK);   // 128 blocks of 64 tokens

  if (ws_size >= w16_bytes) {
    _Float16* w16 = (_Float16*)d_ws;
    cvt_w_kernel<<<dim3(1024), 256, 0, stream>>>(f2w, w16);   // 1M elems / 4 per thread
    vsn_kernel<true><<<grid, 256, 0, stream>>>(x, wg1w, wg1b, wg2w, wg2b, wlng, wlnb,
                                               f1w, f1b, f2w, f2b, fsw, fsb, flng, flnb,
                                               w16, out);
  } else {
    vsn_kernel<false><<<grid, 256, 0, stream>>>(x, wg1w, wg1b, wg2w, wg2b, wlng, wlnb,
                                                f1w, f1b, f2w, f2b, fsw, fsb, flng, flnb,
                                                (const _Float16*)nullptr, out);
  }
}
